// Encoder_55310588838396
// MI455X (gfx1250) — matrix operand order repaired and checked by reading, not compiled, not run on hardware
//
#include <hip/hip_runtime.h>
#include <hip/hip_bf16.h>
#include <stdint.h>
#include <stddef.h>

// Problem constants (match reference)
#define BB    32
#define SS    512
#define DD    512
#define HH    8
#define DKK   64
#define FF    2048
#define LLAY  8
#define MROWS (BB * SS)   // 16384

typedef __bf16 v16bf __attribute__((ext_vector_type(16)));
typedef float  v8f   __attribute__((ext_vector_type(8)));
typedef unsigned int u32x4 __attribute__((ext_vector_type(4)));
typedef int          i32x4 __attribute__((ext_vector_type(4)));
typedef int          i32x8 __attribute__((ext_vector_type(8)));

union FragU { v16bf v; uint4 q[2]; };

// ---- CDNA5 feature detection (device pass only) -----------------------------
#if defined(__HIP_DEVICE_COMPILE__) && defined(__gfx1250__)
#  if __has_builtin(__builtin_amdgcn_tensor_load_to_lds)
#    define USE_TDM 1
#    pragma message("CDNA5 staging path: TDM tensor_load_to_lds")
#  endif
#  if !defined(USE_TDM) && __has_builtin(__builtin_amdgcn_global_load_async_to_lds_b128)
#    define USE_ASYNC 1
#    pragma message("CDNA5 staging path: global_load_async_to_lds_b128")
#  endif
#  if !defined(USE_TDM) && !defined(USE_ASYNC)
#    pragma message("CDNA5 staging path: synchronous fallback (no TDM/async builtin)")
#  endif
#endif
#ifndef USE_TDM
#  define USE_TDM 0
#endif
#ifndef USE_ASYNC
#  define USE_ASYNC 0
#endif

#define AS1 __attribute__((address_space(1)))
#define AS3 __attribute__((address_space(3)))

__device__ __forceinline__ unsigned short f2b(float f) {
    unsigned int u = __float_as_uint(f);
    unsigned int r = u + 0x7FFFu + ((u >> 16) & 1u);   // round-to-nearest-even
    return (unsigned short)(r >> 16);
}

__device__ __forceinline__ v8f wmma_bf16(v16bf a, v16bf b, v8f c) {
    // D(16x16 f32) = A(16x32 bf16) * B(32x16 bf16) + C
    return __builtin_amdgcn_wmma_f32_16x16x32_bf16(
        false, a, false, b, (short)0, c, false, false);
}

#if USE_TDM
__device__ __forceinline__ unsigned lds_addr_of(const void* p) {
    return (unsigned)(unsigned long long)(AS3 const char*)p;
}

// Issue a TDM load of a 2-D tile: 128 rows x 32 bf16, row stride = ldg elements,
// into LDS at lds_addr with 16B padding after each 64B row (LDS stride 80B).
__device__ __forceinline__ void tdm_load_tile(const unsigned short* gtile,
                                              unsigned lds_addr, unsigned ldg) {
    unsigned long long ga = (unsigned long long)(const void*)gtile;  // byte addr
    u32x4 g0;
    g0[0] = 1u;                                            // count=1, user desc
    g0[1] = lds_addr;                                      // LDS byte address
    g0[2] = (unsigned)(ga & 0xFFFFFFFFull);                // global_addr[31:0]
    g0[3] = (unsigned)((ga >> 32) & 0x1FFFFFFull)          // global_addr[56:32]
          | (2u << 30);                                    // type = 2 (image)
    i32x8 g1;
    // data_size=1 (2B), pad_enable, pad_interval=16 DW (code 3), pad_amount=4 DW (code 3)
    g1[0] = (int)((1u << 16) | (1u << 20) | (3u << 22) | (3u << 25));
    const unsigned td0 = 0x40000000u;                      // huge tensor dims (no OOB)
    const unsigned td1 = 0x40000000u;
    g1[1] = (int)((td0 & 0xFFFFu) << 16);                  // tensor_dim0 lo
    g1[2] = (int)(((td0 >> 16) & 0xFFFFu) | ((td1 & 0xFFFFu) << 16));
    g1[3] = (int)(((td1 >> 16) & 0xFFFFu) | (32u << 16));  // tile_dim0 = 32
    g1[4] = (int)(128u);                                   // tile_dim1 = 128, tile_dim2 = 0
    g1[5] = (int)ldg;                                      // tensor_dim0_stride lo32
    g1[6] = 0;                                             // stride hi16 | dim1_stride lo16
    g1[7] = 0;
    i32x4 z4 = (i32x4){0, 0, 0, 0};
#if defined(__clang_major__) && (__clang_major__ >= 23)
    i32x8 z8 = (i32x8){0, 0, 0, 0, 0, 0, 0, 0};
    __builtin_amdgcn_tensor_load_to_lds(g0, g1, z4, z4, z8, 0);
#else
    __builtin_amdgcn_tensor_load_to_lds(g0, g1, z4, z4, 0);
#endif
}
#endif  // USE_TDM

// ---------------------------------------------------------------------------
// f32 [L][K][N] -> bf16 transposed [L][N][K]  (so GEMM B-tiles are row loads)
// ---------------------------------------------------------------------------
__global__ __launch_bounds__(256) void cvt_transpose(const float* __restrict__ W,
                                                     unsigned short* __restrict__ Wt,
                                                     int K, int N, int L) {
    const size_t per = (size_t)K * N;
    const size_t total = per * L;
    for (size_t i = (size_t)blockIdx.x * blockDim.x + threadIdx.x; i < total;
         i += (size_t)gridDim.x * blockDim.x) {
        size_t l = i / per;
        size_t r = i - l * per;
        int k = (int)(r / N);
        int n = (int)(r - (size_t)k * N);
        Wt[l * per + (size_t)n * K + k] = f2b(W[i]);
    }
}

// ---------------------------------------------------------------------------
// Embedding + positional encoding.  One block per (b,s) row, 128 threads x 4.
// ---------------------------------------------------------------------------
__global__ __launch_bounds__(128) void embed_pe(const int* __restrict__ src,
                                                const float* __restrict__ emb,
                                                float* __restrict__ x,
                                                unsigned short* __restrict__ xb) {
    const int bs  = blockIdx.x;
    const int s   = bs & (SS - 1);
    const int tok = src[bs];
    const size_t ebase = (size_t)tok * DD;
    const size_t obase = (size_t)bs * DD;
    const float sqrtD = 22.627416997969522f;            // sqrt(512)
    const float c = -9.210340371976184f / (float)DD;    // -ln(10000)/D
    int d0 = threadIdx.x * 4;
#pragma unroll
    for (int j = 0; j < 4; ++j) {
        int d  = d0 + j;
        int i2 = d & ~1;
        float ang = (float)s * __expf((float)i2 * c);
        float pe  = (d & 1) ? __cosf(ang) : __sinf(ang);
        float v   = emb[ebase + d] * sqrtD + pe;
        x[obase + d]  = v;
        xb[obase + d] = f2b(v);
    }
}

// ---------------------------------------------------------------------------
// bf16 GEMM:  out[M x N] = A[M x K] @ Wt[N x K]^T + bias
// MODE: 0 = f32 out (non-temporal), 1 = bf16 out, 2 = bf16 out + relu,
//       3 = bf16 out in attention-V transposed layout [b,h,dk,s]
// Block tile 128x128, BK=32, 256 threads = 8 waves (4x2).
// Double-buffered LDS with an explicitly 2x-unrolled K loop (buffer indices are
// compile-time constants).  Tiles staged by TDM when available (A by wave0,
// B by wave1), else async-to-LDS, else synchronous loads.
// Requires K % 64 == 0 (true here: K is 512 or 2048).
// ---------------------------------------------------------------------------
#define GEMM_LDA 40   // padded LDS row stride (elements): 64B data + 16B pad

template <int MODE>
__global__ __launch_bounds__(256) void gemm_bf16(const unsigned short* __restrict__ A,
                                                 const unsigned short* __restrict__ Wt,
                                                 const float* __restrict__ bias,
                                                 float* __restrict__ outF,
                                                 unsigned short* __restrict__ outB,
                                                 int M, int N, int K) {
    __shared__ __align__(16) unsigned short sA[2][128 * GEMM_LDA];
    __shared__ __align__(16) unsigned short sB[2][128 * GEMM_LDA];

    const int tid   = threadIdx.x;
    const int wave  = tid >> 5;
    const int lane  = tid & 31;
    const int half  = lane >> 4;
    const int l15   = lane & 15;
    const int mW    = (wave >> 1) * 32;
    const int nW    = (wave & 1) * 64;
    const int bm    = blockIdx.y * 128;
    const int bn    = blockIdx.x * 128;

    const unsigned short* Atile = A  + (size_t)bm * K;   // + k0
    const unsigned short* Btile = Wt + (size_t)bn * K;   // + k0

    v8f acc[2][4];
#pragma unroll
    for (int mi = 0; mi < 2; ++mi)
#pragma unroll
        for (int ni = 0; ni < 4; ++ni)
            acc[mi][ni] = (v8f){0.f, 0.f, 0.f, 0.f, 0.f, 0.f, 0.f, 0.f};

    // ---- staging of one 128x32 k-step into LDS buffer `buf` ----
    auto stage = [&](int buf, int kstep) {
        const int k0 = kstep << 5;
#if USE_TDM
        if (tid == 0)
            tdm_load_tile(Atile + k0, lds_addr_of(&sA[buf][0]), (unsigned)K);
        else if (tid == 32)
            tdm_load_tile(Btile + k0, lds_addr_of(&sB[buf][0]), (unsigned)K);
#else
#pragma unroll
        for (int it = 0; it < 2; ++it) {
            int idx = tid * 8 + it * 2048;
            int r = idx >> 5, c = idx & 31;
            const unsigned short* ga = Atile + (size_t)r * K + k0 + c;
            const unsigned short* gb = Btile + (size_t)r * K + k0 + c;
            unsigned short* la = &sA[buf][r * GEMM_LDA + c];
            unsigned short* lb = &sB[buf][r * GEMM_LDA + c];
#if USE_ASYNC
            __builtin_amdgcn_global_load_async_to_lds_b128(
                (AS1 const void*)ga, (AS3 void*)la, 0, 0);
            __builtin_amdgcn_global_load_async_to_lds_b128(
                (AS1 const void*)gb, (AS3 void*)lb, 0, 0);
#else
            *(uint4*)la = *(const uint4*)ga;
            *(uint4*)lb = *(const uint4*)gb;
#endif
        }
#endif
    };

    auto wait_staging = [&]() {
#if USE_TDM
        if (lane == 0 && wave < 2) __builtin_amdgcn_s_wait_tensorcnt(0);
#elif USE_ASYNC
#  if __has_builtin(__builtin_amdgcn_s_wait_asynccnt)
        __builtin_amdgcn_s_wait_asynccnt(0);
#  else
        asm volatile("s_wait_asynccnt 0" ::: "memory");
#  endif
#endif
    };

    // ---- one K-step of WMMA work from LDS buffer `buf` (compile-time const) ----
    auto compute = [&](int buf) {
        FragU fa[2];
#pragma unroll
        for (int mi = 0; mi < 2; ++mi) {
            const unsigned short* pr = &sA[buf][(mW + mi * 16 + l15) * GEMM_LDA];
            fa[mi].q[0] = *(const uint4*)(pr + half * 8);
            fa[mi].q[1] = *(const uint4*)(pr + 16 + half * 8);
        }
        FragU fb[4];
#pragma unroll
        for (int ni = 0; ni < 4; ++ni) {
            const unsigned short* pc =
                &sB[buf][(nW + ni * 16 + l15) * GEMM_LDA + half * 8];
            fb[ni].q[0] = *(const uint4*)(pc);
            fb[ni].q[1] = *(const uint4*)(pc + 16);
        }
#pragma unroll
        for (int mi = 0; mi < 2; ++mi)
#pragma unroll
            for (int ni = 0; ni < 4; ++ni)
                acc[mi][ni] = wmma_bf16(fa[mi].v, fb[ni].v, acc[mi][ni]);
    };

    const int nK = K >> 5;        // even (K % 64 == 0)

    stage(0, 0);                  // prologue
    for (int kt = 0; kt < nK; kt += 2) {
        // even step: consume buffer 0, prefetch into buffer 1
        wait_staging();
        __syncthreads();
        stage(1, kt + 1);         // kt+1 < nK because nK is even
        compute(0);
        // odd step: consume buffer 1, prefetch into buffer 0
        wait_staging();
        __syncthreads();
        if (kt + 2 < nK) stage(0, kt + 2);
        compute(1);
    }

    // ---- epilogue (branch-free via MODE) ----
#pragma unroll
    for (int mi = 0; mi < 2; ++mi) {
#pragma unroll
        for (int ni = 0; ni < 4; ++ni) {
            const int col = bn + nW + ni * 16 + l15;
            const float bv = bias[col];
#pragma unroll
            for (int r = 0; r < 8; ++r) {
                const int row = bm + mW + mi * 16 + r + 8 * half;
                float v = acc[mi][ni][r] + bv;
                if constexpr (MODE == 2) v = fmaxf(v, 0.f);
                if constexpr (MODE == 0) {
                    // streamed once into add_ln; use non-temporal hint
                    __builtin_nontemporal_store(v, outF + (size_t)row * N + col);
                } else if constexpr (MODE == 3) {
                    // [b, h, dk, s] layout for attention V
                    const int b_ = row >> 9, s_ = row & (SS - 1);
                    const int h_ = col >> 6, d_ = col & (DKK - 1);
                    const size_t o =
                        ((((size_t)b_ * HH + h_) * DKK + d_) << 9) + s_;
                    outB[o] = f2b(v);
                } else {
                    outB[(size_t)row * N + col] = f2b(v);
                }
            }
        }
    }
}

// ---------------------------------------------------------------------------
// Fused attention for one (b, h, 16-row q tile).  128 threads = 4 waves.
//   scores = (Q K^T) * dk^-0.5  (WMMA, operands direct from global)
//   masked softmax in LDS
//   O = P V   (WMMA; V pre-transposed [b,h,dk,s] so B-frags are contiguous)
// ---------------------------------------------------------------------------
#define P_LD 520

__global__ __launch_bounds__(128) void attention(const unsigned short* __restrict__ Qb,
                                                 const unsigned short* __restrict__ Kb,
                                                 const unsigned short* __restrict__ Vt,
                                                 const int* __restrict__ mask,
                                                 unsigned short* __restrict__ Ob) {
    __shared__ float          sS[16 * SS];     // 32 KB scores
    __shared__ unsigned short sP[16 * P_LD];   // 16.25 KB probabilities (bf16)

    const int tid  = threadIdx.x;
    const int wave = tid >> 5;
    const int lane = tid & 31;
    const int half = lane >> 4;
    const int l15  = lane & 15;
    const int q0   = blockIdx.x * 16;
    const int h    = blockIdx.y;
    const int b    = blockIdx.z;
    const size_t rowbase = (size_t)b * SS;

    // ---- preload Q fragments (invariant over key tiles) ----
    FragU fq[2];
    {
        const unsigned short* pq = Qb + (rowbase + q0 + l15) * DD + h * DKK;
#pragma unroll
        for (int kc = 0; kc < 2; ++kc) {
            fq[kc].q[0] = *(const uint4*)(pq + kc * 32 + half * 8);
            fq[kc].q[1] = *(const uint4*)(pq + kc * 32 + 16 + half * 8);
        }
    }

    // ---- phase 1: scores = Q @ K^T, each wave covers 128 keys ----
    for (int nt = 0; nt < 8; ++nt) {
        const int n0 = wave * 128 + nt * 16;
        v8f acc = (v8f){0.f, 0.f, 0.f, 0.f, 0.f, 0.f, 0.f, 0.f};
#pragma unroll
        for (int kc = 0; kc < 2; ++kc) {
            FragU fb;
            const unsigned short* pk =
                Kb + (rowbase + n0 + l15) * DD + h * DKK + kc * 32 + half * 8;
            fb.q[0] = *(const uint4*)(pk);
            fb.q[1] = *(const uint4*)(pk + 16);
            acc = wmma_bf16(fq[kc].v, fb.v, acc);
        }
#pragma unroll
        for (int r = 0; r < 8; ++r)
            sS[(r + 8 * half) * SS + n0 + l15] = acc[r] * 0.125f;   // dk^-0.5
    }
    __syncthreads();

    // ---- phase 2: masked softmax, 8 lanes cooperate per row ----
    {
        const int row = tid >> 3;
        const int sub = tid & 7;
        const int* mrow = mask + rowbase;
        float mx = -1e30f;
        for (int j = 0; j < 64; ++j) {
            const int col = sub + (j << 3);
            float sv = sS[row * SS + col];
            sv = (mrow[col] != 0) ? sv : -1e30f;
            sS[row * SS + col] = sv;
            mx = fmaxf(mx, sv);
        }
        for (int o = 1; o < 8; o <<= 1) mx = fmaxf(mx, __shfl_xor(mx, o, 32));
        float sum = 0.f;
        for (int j = 0; j < 64; ++j) {
            const int col = sub + (j << 3);
            float p = __expf(sS[row * SS + col] - mx);
            sS[row * SS + col] = p;
            sum += p;
        }
        for (int o = 1; o < 8; o <<= 1) sum += __shfl_xor(sum, o, 32);
        const float rinv = 1.f / fmaxf(sum, 1e-20f);
        for (int j = 0; j < 64; ++j) {
            const int col = sub + (j << 3);
            sP[row * P_LD + col] = f2b(sS[row * SS + col] * rinv);
        }
    }
    __syncthreads();

    // ---- phase 3: O = P @ V, each wave owns 16 head dims ----
    v8f accO = (v8f){0.f, 0.f, 0.f, 0.f, 0.f, 0.f, 0.f, 0.f};
    const unsigned short* pvbase =
        Vt + ((size_t)(b * HH + h) * DKK + wave * 16 + l15) * SS + half * 8;
    const unsigned short* ppbase = sP + l15 * P_LD;
    for (int kc = 0; kc < 16; ++kc) {
        FragU fa, fb;
        const unsigned short* pp = ppbase + kc * 32;
        fa.q[0] = *(const uint4*)(pp + half * 8);
        fa.q[1] = *(const uint4*)(pp + 16 + half * 8);
        const unsigned short* pv = pvbase + kc * 32;
        fb.q[0] = *(const uint4*)(pv);
        fb.q[1] = *(const uint4*)(pv + 16);
        accO = wmma_bf16(fa.v, fb.v, accO);
    }
#pragma unroll
    for (int r = 0; r < 8; ++r) {
        const int qr = q0 + r + 8 * half;
        Ob[(rowbase + qr) * DD + h * DKK + wave * 16 + l15] = f2b(accO[r]);
    }
}

// ---------------------------------------------------------------------------
// Residual add + LayerNorm.  One block per row, 128 threads x 4 elements.
// ---------------------------------------------------------------------------
__global__ __launch_bounds__(128) void add_ln(const float* __restrict__ x_old,
                                              const float* __restrict__ delta,
                                              const float* __restrict__ g,
                                              const float* __restrict__ bt,
                                              float* __restrict__ x_out,
                                              unsigned short* __restrict__ xb_out) {
    __shared__ float red[8];
    const int tid  = threadIdx.x;
    const int wave = tid >> 5;
    const int lane = tid & 31;
    const size_t base = (size_t)blockIdx.x * DD;
    const int d0 = tid * 4;

    float4 a  = *(const float4*)(x_old + base + d0);
    float4 dd = *(const float4*)(delta + base + d0);
    float v0 = a.x + dd.x, v1 = a.y + dd.y, v2 = a.z + dd.z, v3 = a.w + dd.w;
    float s  = v0 + v1 + v2 + v3;
    float s2 = v0 * v0 + v1 * v1 + v2 * v2 + v3 * v3;
    for (int o = 16; o > 0; o >>= 1) {
        s  += __shfl_xor(s, o, 32);
        s2 += __shfl_xor(s2, o, 32);
    }
    if (lane == 0) { red[wave] = s; red[4 + wave] = s2; }
    __syncthreads();
    s  = red[0] + red[1] + red[2] + red[3];
    s2 = red[4] + red[5] + red[6] + red[7];
    const float mean = s * (1.f / DD);
    const float var  = s2 * (1.f / DD) - mean * mean;
    const float rstd = rsqrtf(var + 1e-5f);

    float vv[4] = {v0, v1, v2, v3};
#pragma unroll
    for (int j = 0; j < 4; ++j) {
        const int d = d0 + j;
        const float y = (vv[j] - mean) * rstd * g[d] + bt[d];
        x_out[base + d]  = y;
        xb_out[base + d] = f2b(y);
    }
}

// ---------------------------------------------------------------------------
// Host-side orchestration
// ---------------------------------------------------------------------------
extern "C" void kernel_launch(void* const* d_in, const int* in_sizes, int n_in,
                              void* d_out, int out_size, void* d_ws, size_t ws_size,
                              hipStream_t stream) {
    (void)in_sizes; (void)n_in; (void)ws_size;

    const int*   src  = (const int*)d_in[0];
    const int*   mask = (const int*)d_in[1];
    const float* emb  = (const float*)d_in[2];
    const float* Wq   = (const float*)d_in[3];
    const float* bq   = (const float*)d_in[4];
    const float* Wk   = (const float*)d_in[5];
    const float* bk   = (const float*)d_in[6];
    const float* Wv   = (const float*)d_in[7];
    const float* bv   = (const float*)d_in[8];
    const float* Wo   = (const float*)d_in[9];
    const float* bo   = (const float*)d_in[10];
    const float* W1   = (const float*)d_in[11];
    const float* b1   = (const float*)d_in[12];
    const float* W2   = (const float*)d_in[13];
    const float* b2   = (const float*)d_in[14];
    const float* ln1g = (const float*)d_in[15];
    const float* ln1b = (const float*)d_in[16];
    const float* ln2g = (const float*)d_in[17];
    const float* ln2b = (const float*)d_in[18];

    char* w = (char*)d_ws;
    auto take = [&](size_t bytes) -> void* {
        void* p = (void*)w;
        w += (bytes + 255) & ~(size_t)255;
        return p;
    };
    const size_t LDD = (size_t)LLAY * DD * DD;
    const size_t LDF = (size_t)LLAY * DD * FF;

    unsigned short* wqt = (unsigned short*)take(LDD * 2);   // [L][N=D][K=D]
    unsigned short* wkt = (unsigned short*)take(LDD * 2);
    unsigned short* wvt = (unsigned short*)take(LDD * 2);
    unsigned short* wot = (unsigned short*)take(LDD * 2);
    unsigned short* w1t = (unsigned short*)take(LDF * 2);   // [L][N=F][K=D]
    unsigned short* w2t = (unsigned short*)take(LDF * 2);   // [L][N=D][K=F]
    float*          x   = (float*)take((size_t)MROWS * DD * 4);
    unsigned short* xb  = (unsigned short*)take((size_t)MROWS * DD * 2);
    unsigned short* qb  = (unsigned short*)take((size_t)MROWS * DD * 2);
    unsigned short* kbf = (unsigned short*)take((size_t)MROWS * DD * 2);
    unsigned short* vtb = (unsigned short*)take((size_t)MROWS * DD * 2); // [b,h,dk,s]
    unsigned short* ab  = (unsigned short*)take((size_t)MROWS * DD * 2);
    unsigned short* h1b = (unsigned short*)take((size_t)MROWS * FF * 2);
    float*          tmp = (float*)take((size_t)MROWS * DD * 4);

    // weights -> bf16, transposed to [N][K]
    cvt_transpose<<<4096, 256, 0, stream>>>(Wq, wqt, DD, DD, LLAY);
    cvt_transpose<<<4096, 256, 0, stream>>>(Wk, wkt, DD, DD, LLAY);
    cvt_transpose<<<4096, 256, 0, stream>>>(Wv, wvt, DD, DD, LLAY);
    cvt_transpose<<<4096, 256, 0, stream>>>(Wo, wot, DD, DD, LLAY);
    cvt_transpose<<<8192, 256, 0, stream>>>(W1, w1t, DD, FF, LLAY);
    cvt_transpose<<<8192, 256, 0, stream>>>(W2, w2t, FF, DD, LLAY);

    embed_pe<<<MROWS, 128, 0, stream>>>(src, emb, x, xb);

    const dim3 gD(DD / 128, MROWS / 128);   // N=512 GEMMs
    const dim3 gF(FF / 128, MROWS / 128);   // N=2048 GEMM
    const dim3 gAtt(SS / 16, HH, BB);

    for (int l = 0; l < LLAY; ++l) {
        const size_t wOff  = (size_t)l * DD * DD;
        const size_t bOff  = (size_t)l * DD;
        const size_t w1Off = (size_t)l * DD * FF;
        const size_t b1Off = (size_t)l * FF;

        gemm_bf16<1><<<gD, 256, 0, stream>>>(xb, wqt + wOff, bq + bOff,
                                             nullptr, qb, MROWS, DD, DD);
        gemm_bf16<1><<<gD, 256, 0, stream>>>(xb, wkt + wOff, bk + bOff,
                                             nullptr, kbf, MROWS, DD, DD);
        gemm_bf16<3><<<gD, 256, 0, stream>>>(xb, wvt + wOff, bv + bOff,
                                             nullptr, vtb, MROWS, DD, DD);

        attention<<<gAtt, 128, 0, stream>>>(qb, kbf, vtb, mask, ab);

        gemm_bf16<0><<<gD, 256, 0, stream>>>(ab, wot + wOff, bo + bOff,
                                             tmp, nullptr, MROWS, DD, DD);
        add_ln<<<MROWS, 128, 0, stream>>>(x, tmp, ln1g + bOff, ln1b + bOff, x, xb);

        gemm_bf16<2><<<gF, 256, 0, stream>>>(xb, w1t + w1Off, b1 + b1Off,
                                             nullptr, h1b, MROWS, FF, DD);
        gemm_bf16<0><<<gD, 256, 0, stream>>>(h1b, w2t + w1Off, b2 + bOff,
                                             tmp, nullptr, MROWS, DD, FF);
        add_ln<<<MROWS, 128, 0, stream>>>(x, tmp, ln2g + bOff, ln2b + bOff, x, xb);
    }

    hipMemcpyAsync(d_out, x, (size_t)out_size * sizeof(float),
                   hipMemcpyDeviceToDevice, stream);
}
